// MoEGPT_26439818674681
// MI455X (gfx1250) — matrix operand order repaired and checked by reading, not compiled, not run on hardware
//
#include <hip/hip_runtime.h>
#include <hip/hip_bf16.h>

// ---- model dims --------------------------------------------------------
#define CV 32000
#define CC 1024
#define C3 3072
#define CH 16
#define CL 3
#define CE 8
#define CD 1024
#define CB 8
#define CT 512
#define CFF 4096
#define CN 4096   // B*T

typedef __attribute__((ext_vector_type(16))) __bf16 v16bf;
typedef __attribute__((ext_vector_type(8)))  float  v8f;
typedef int v4i __attribute__((vector_size(16)));

struct alignas(16) U128 { unsigned int x, y, z, w; };
union FragBF { U128 u[2]; v16bf v; };

__device__ __forceinline__ unsigned short f2bf(float f) {
  unsigned int u = __float_as_uint(f);
  u += 0x7fffu + ((u >> 16) & 1u);          // round-to-nearest-even
  return (unsigned short)(u >> 16);
}

__device__ __forceinline__ float gelu_exact(float x) {
  return 0.5f * x * (1.0f + erff(x * 0.70710678118654752f));
}

enum { GF_BIAS = 1, GF_GELU = 2, GF_RES = 4, GF_SCALE = 8,
       GF_ACC = 16, GF_OUTF = 32, GF_OUTB = 64 };

// ---- CDNA5 async global->LDS helpers -----------------------------------
#if __has_builtin(__builtin_amdgcn_global_load_async_to_lds_b128)
#define USE_ASYNC_LDS 1
#else
#define USE_ASYNC_LDS 0
#endif

__device__ __forceinline__ void async_ld16(const unsigned short* g, unsigned short* l) {
#if USE_ASYNC_LDS
  __builtin_amdgcn_global_load_async_to_lds_b128(
      (__attribute__((address_space(1))) v4i*)g,
      (__attribute__((address_space(3))) v4i*)l, 0, 0);
#else
  *(U128*)l = *(const U128*)g;
#endif
}

template <int N>
__device__ __forceinline__ void wait_async() {
#if __has_builtin(__builtin_amdgcn_s_wait_asynccnt)
  __builtin_amdgcn_s_wait_asynccnt(N);
#else
  asm volatile("s_wait_asynccnt %0" :: "n"(N) : "memory");
#endif
}

// ---- generic bf16 WMMA GEMM:  C[M,N] = A[M,K] * Bt[N,K]^T (+epilogue) --
// Block: 256 thr = 8 waves, tile 128x128, K-chunk 32, double-buffered LDS
// filled by async global->LDS loads (ASYNCcnt pipeline).
// Wave (wm,wn) in 2x4 grid computes 64x32 = 4x2 wmma 16x16 tiles.
__global__ __launch_bounds__(256) void k_gemm_bf16_wmma(
    const unsigned short* __restrict__ A,   // [M,K] bf16 bits
    const unsigned short* __restrict__ Bt,  // [N,K] bf16 bits (B transposed)
    const float* __restrict__ bias,         // [N] or null
    const float* __restrict__ res,          // [M,N] or null
    const float* __restrict__ rowScale,     // [M] or null
    float* __restrict__ Cf,
    unsigned short* __restrict__ Cb,
    int M, int N, int K, int flags)
{
  constexpr int LDT = 40;                   // padded halfs/row (80B stride)
  __shared__ __align__(16) unsigned short lsA[2][128 * LDT];
  __shared__ __align__(16) unsigned short lsB[2][128 * LDT];

  const int tid  = threadIdx.x;
  const int lane = tid & 31;
  const int wave = tid >> 5;
  const int wm = wave >> 2;                 // 0..1
  const int wn = wave & 3;                  // 0..3
  const int bm = blockIdx.y * 128;
  const int bn = blockIdx.x * 128;

  v8f acc[4][2];
  for (int i = 0; i < 4; i++)
    for (int j = 0; j < 2; j++)
      for (int r = 0; r < 8; r++) acc[i][j][r] = 0.f;

  const int lr = tid >> 2;                  // 0..63 (tile row for loads)
  const int lc = (tid & 3) * 8;             // 8-half chunk

  const int khalf = lane >> 4;              // 0..1
  const int rr    = lane & 15;

  // 4 async-load instructions per thread per tile => ASYNCcnt +4 per wave
  auto issue_tile = [&](int k0, int bufi) {
    async_ld16(&A[(size_t)(bm + lr) * K + k0 + lc],       &lsA[bufi][lr * LDT + lc]);
    async_ld16(&A[(size_t)(bm + lr + 64) * K + k0 + lc],  &lsA[bufi][(lr + 64) * LDT + lc]);
    async_ld16(&Bt[(size_t)(bn + lr) * K + k0 + lc],      &lsB[bufi][lr * LDT + lc]);
    async_ld16(&Bt[(size_t)(bn + lr + 64) * K + k0 + lc], &lsB[bufi][(lr + 64) * LDT + lc]);
  };

  issue_tile(0, 0);
  const int ntile = K >> 5;
  int buf = 0;
  for (int it = 0; it < ntile; ++it) {
    if (it + 1 < ntile) {                   // prefetch next tile into spare buffer
      issue_tile((it + 1) << 5, buf ^ 1);
      wait_async<4>();                      // drain only current tile's 4 loads
    } else {
      wait_async<0>();
    }
    __syncthreads();

    // B fragment: lane holds column (rr); halfs = K[khalf*8..+8) and K[16+khalf*8..+8)
    FragBF fb[2];
#pragma unroll
    for (int nt = 0; nt < 2; nt++) {
      int nb = wn * 32 + nt * 16 + rr;
      fb[nt].u[0] = *(const U128*)(&lsB[buf][nb * LDT + khalf * 8]);
      fb[nt].u[1] = *(const U128*)(&lsB[buf][nb * LDT + 16 + khalf * 8]);
    }
    // A fragment: lane holds row (rr); halfs = K[khalf*8..+8) and K[16+khalf*8..+8)
#pragma unroll
    for (int mt = 0; mt < 4; mt++) {
      int mb = wm * 64 + mt * 16 + rr;
      FragBF fa;
      fa.u[0] = *(const U128*)(&lsA[buf][mb * LDT + khalf * 8]);
      fa.u[1] = *(const U128*)(&lsA[buf][mb * LDT + 16 + khalf * 8]);
#pragma unroll
      for (int nt = 0; nt < 2; nt++) {
        acc[mt][nt] = __builtin_amdgcn_wmma_f32_16x16x32_bf16(
            false, fa.v, false, fb[nt].v, (short)0, acc[mt][nt], false, false);
      }
    }
    __syncthreads();                        // all reads of buf done before reuse
    buf ^= 1;
  }

  // epilogue; C/D layout: vgpr r holds M = r + 8*(lane>>4), col = lane&15
  const int rowhi = khalf * 8;
#pragma unroll
  for (int mt = 0; mt < 4; mt++) {
#pragma unroll
    for (int nt = 0; nt < 2; nt++) {
#pragma unroll
      for (int r = 0; r < 8; r++) {
        int row = bm + wm * 64 + mt * 16 + rowhi + r;
        int col = bn + wn * 32 + nt * 16 + rr;
        float v = acc[mt][nt][r];
        if (flags & GF_BIAS)  v += bias[col];
        if (flags & GF_GELU)  v  = gelu_exact(v);
        if (flags & GF_SCALE) v *= rowScale[row];
        size_t o = (size_t)row * N + col;
        if (flags & GF_RES)   v += res[o];
        if (flags & GF_ACC)   v += Cf[o];
        if (flags & GF_OUTF)  Cf[o] = v;
        if (flags & GF_OUTB)  Cb[o] = f2bf(v);
      }
    }
  }
}

// ---- transpose + fp32->bf16 cast:  src[K,N] -> dst[N,K] ----------------
__global__ __launch_bounds__(256) void k_transp_bf16(
    const float* __restrict__ src, unsigned short* __restrict__ dst, int K, int N)
{
  __shared__ float tile[32][33];
  int n0 = blockIdx.x * 32, k0 = blockIdx.y * 32;
  int c = threadIdx.x & 31, r8 = threadIdx.x >> 5;
#pragma unroll
  for (int i = 0; i < 4; i++) {
    int r = r8 + i * 8;
    tile[r][c] = src[(size_t)(k0 + r) * N + n0 + c];
  }
  __syncthreads();
#pragma unroll
  for (int i = 0; i < 4; i++) {
    int r = r8 + i * 8;
    dst[(size_t)(n0 + r) * K + k0 + c] = f2bf(tile[c][r]);
  }
}

__global__ __launch_bounds__(256) void k_cvt_bf16(
    const float* __restrict__ s, unsigned short* __restrict__ d, int n)
{
  int i = blockIdx.x * 256 + threadIdx.x;
  if (i < n) d[i] = f2bf(s[i]);
}

// ---- embedding gather ---------------------------------------------------
__global__ __launch_bounds__(256) void k_embed(
    const int* __restrict__ idx, const float* __restrict__ emb, float* __restrict__ x)
{
  int n = blockIdx.x;
  int tok = idx[n];
  const float* e = emb + (size_t)tok * CC;
  for (int c = threadIdx.x; c < CC; c += 256) x[(size_t)n * CC + c] = e[c];
}

// ---- layernorm: fp32 in, bf16 out (one block per row) ------------------
__global__ __launch_bounds__(256) void k_layernorm(
    const float* __restrict__ x, const float* __restrict__ g,
    const float* __restrict__ b, unsigned short* __restrict__ out)
{
  int row = blockIdx.x;
  const float* xr = x + (size_t)row * CC;
  float s = 0.f, s2 = 0.f;
  for (int c = threadIdx.x; c < CC; c += 256) { float v = xr[c]; s += v; s2 += v * v; }
  for (int o = 16; o; o >>= 1) { s += __shfl_xor(s, o); s2 += __shfl_xor(s2, o); }
  __shared__ float red[2][8];
  __shared__ float bc[2];
  int lane = threadIdx.x & 31, wave = threadIdx.x >> 5;
  if (lane == 0) { red[0][wave] = s; red[1][wave] = s2; }
  __syncthreads();
  if (wave == 0) {
    float a  = lane < 8 ? red[0][lane] : 0.f;
    float a2 = lane < 8 ? red[1][lane] : 0.f;
    for (int o = 4; o; o >>= 1) { a += __shfl_xor(a, o); a2 += __shfl_xor(a2, o); }
    if (lane == 0) {
      float mean = a * (1.0f / CC);
      float var  = a2 * (1.0f / CC) - mean * mean;
      bc[0] = mean; bc[1] = rsqrtf(var + 1e-5f);
    }
  }
  __syncthreads();
  float mean = bc[0], inv = bc[1];
  for (int c = threadIdx.x; c < CC; c += 256)
    out[(size_t)row * CC + c] = f2bf((xr[c] - mean) * inv * g[c] + b[c]);
}

// ---- RoPE + QKV split: qkv[N,3C] -> Q/K/V [B,H,T,64] -------------------
__global__ __launch_bounds__(256) void k_rope_split(
    const float* __restrict__ qkv, float* __restrict__ Q,
    float* __restrict__ Kd, float* __restrict__ Vd)
{
  int n = blockIdx.x;
  int b = n / CT, t = n % CT;
  const float* base = qkv + (size_t)n * C3;
  for (int c = threadIdx.x; c < CC; c += 256) {
    int h = c >> 6, dd = c & 63, fi = dd & 31;
    float inv = __expf(-(float)(2 * fi) * (9.210340371976184f / 64.f)); // 10000^(-2fi/64)
    float ang = (float)t * inv;
    float cs = __cosf(ang), sn = __sinf(ang);
    float q = base[c], k = base[CC + c], v = base[2 * CC + c];
    float qr = (dd < 32) ? -base[c + 32]      : base[c - 32];
    float kr = (dd < 32) ? -base[CC + c + 32] : base[CC + c - 32];
    size_t o = ((size_t)(b * CH + h) * CT + t) * 64 + dd;
    Q[o]  = q * cs + qr * sn;
    Kd[o] = k * cs + kr * sn;
    Vd[o] = v;
  }
}

// ---- causal attention: wave per query row; writes y as bf16 [N,C] ------
__global__ __launch_bounds__(256) void k_attn(
    const float* __restrict__ Q, const float* __restrict__ Kd,
    const float* __restrict__ Vd, unsigned short* __restrict__ yout)
{
  int rb = blockIdx.x % (CT / 8);
  int bh = blockIdx.x / (CT / 8);
  int wave = threadIdx.x >> 5, lane = threadIdx.x & 31;
  int qi = rb * 8 + wave;

  __shared__ float sq[8][64];
  __shared__ float sp[8][CT];

  const float* qrow = Q + ((size_t)bh * CT + qi) * 64;
  sq[wave][lane]      = qrow[lane];
  sq[wave][lane + 32] = qrow[lane + 32];

  const float scale = 0.125f;               // 1/sqrt(64)
  int nj = (qi + 32) >> 5;                  // ceil((qi+1)/32) key chunks
  float mx = -1e30f;
  for (int jt = 0; jt < nj; jt++) {
    int j = jt * 32 + lane;
    float s = -1e30f;
    if (j <= qi) {
      const float* kr = Kd + ((size_t)bh * CT + j) * 64;
      float a = 0.f;
#pragma unroll
      for (int dd = 0; dd < 64; dd++) a += sq[wave][dd] * kr[dd];
      s = a * scale;
    }
    sp[wave][jt * 32 + lane] = s;
    mx = fmaxf(mx, s);
  }
  for (int o = 16; o; o >>= 1) mx = fmaxf(mx, __shfl_xor(mx, o));
  float sum = 0.f;
  for (int jt = 0; jt < nj; jt++) {
    float e = __expf(sp[wave][jt * 32 + lane] - mx);
    sp[wave][jt * 32 + lane] = e;
    sum += e;
  }
  for (int o = 16; o; o >>= 1) sum += __shfl_xor(sum, o);
  float inv = 1.0f / sum;

  float y0 = 0.f, y1 = 0.f;
  int jmax = nj * 32;
  for (int j = 0; j < jmax; j++) {
    float p = sp[wave][j];
    const float* vr = Vd + ((size_t)bh * CT + j) * 64;
    y0 += p * vr[lane];
    y1 += p * vr[lane + 32];
  }
  int b = bh / CH, h = bh % CH;
  size_t orow = (size_t)(b * CT + qi) * CC + h * 64;
  yout[orow + lane]      = f2bf(y0 * inv);
  yout[orow + lane + 32] = f2bf(y1 * inv);
}

// ---- router: softmax -> top2 -> renorm -> dense rowScale[E,N] ----------
__global__ __launch_bounds__(256) void k_router(
    const float* __restrict__ x, const float* __restrict__ Wr, float* __restrict__ rsc)
{
  int lane = threadIdx.x & 31, wave = threadIdx.x >> 5;
  int n = blockIdx.x * 8 + wave;
  float a[CE];
#pragma unroll
  for (int e = 0; e < CE; e++) a[e] = 0.f;
  for (int c = lane; c < CC; c += 32) {
    float xv = x[(size_t)n * CC + c];
    const float* w = Wr + (size_t)c * CE;
#pragma unroll
    for (int e = 0; e < CE; e++) a[e] += xv * w[e];
  }
#pragma unroll
  for (int e = 0; e < CE; e++)
    for (int o = 16; o; o >>= 1) a[e] += __shfl_xor(a[e], o);
  float m = a[0];
#pragma unroll
  for (int e = 1; e < CE; e++) m = fmaxf(m, a[e]);
  float p[CE];
#pragma unroll
  for (int e = 0; e < CE; e++) p[e] = __expf(a[e] - m);
  int e0 = 0;
#pragma unroll
  for (int e = 1; e < CE; e++) if (p[e] > p[e0]) e0 = e;
  int e1 = (e0 == 0) ? 1 : 0;
#pragma unroll
  for (int e = 0; e < CE; e++) if (e != e0 && p[e] > p[e1]) e1 = e;
  float denom = p[e0] + p[e1];
  float w0 = p[e0] / denom, w1 = p[e1] / denom;
  if (lane < CE)
    rsc[(size_t)lane * CN + n] = (lane == e0) ? w0 : ((lane == e1) ? w1 : 0.f);
}

// ---- host orchestration -------------------------------------------------
extern "C" void kernel_launch(void* const* d_in, const int* in_sizes, int n_in,
                              void* d_out, int out_size, void* d_ws, size_t ws_size,
                              hipStream_t stream)
{
  (void)in_sizes; (void)n_in; (void)out_size; (void)ws_size;
  const int*   idx      = (const int*)  d_in[0];
  const float* tok_emb  = (const float*)d_in[1];
  const float* ln1_g    = (const float*)d_in[2];
  const float* ln1_b    = (const float*)d_in[3];
  const float* ln2_g    = (const float*)d_in[4];
  const float* ln2_b    = (const float*)d_in[5];
  const float* qkv_w    = (const float*)d_in[6];
  const float* proj_w   = (const float*)d_in[7];
  const float* ff_w1    = (const float*)d_in[8];
  const float* ff_b1    = (const float*)d_in[9];
  const float* ff_w2    = (const float*)d_in[10];
  const float* ff_b2    = (const float*)d_in[11];
  const float* router_w = (const float*)d_in[12];
  const float* exp_w1   = (const float*)d_in[13];
  const float* exp_b1   = (const float*)d_in[14];
  const float* exp_w2   = (const float*)d_in[15];
  const float* exp_b2   = (const float*)d_in[16];
  const float* lnf_g    = (const float*)d_in[17];
  const float* lnf_b    = (const float*)d_in[18];
  const float* head_w   = (const float*)d_in[19];
  float* logits = (float*)d_out;

  // bump allocator over d_ws
  char* base = (char*)d_ws; size_t off = 0;
  auto take = [&](size_t bytes) -> void* {
    void* p = base + off; off = (off + bytes + 255) & ~(size_t)255; return p;
  };
  unsigned short* qkv_wt = (unsigned short*)take((size_t)CL * C3 * CC * 2);
  unsigned short* proj_wt= (unsigned short*)take((size_t)CL * CC * CC * 2);
  unsigned short* ff1_wt = (unsigned short*)take((size_t)CL * CFF * CC * 2);
  unsigned short* ff2_wt = (unsigned short*)take((size_t)CL * CC * CFF * 2);
  unsigned short* ex1_wt = (unsigned short*)take((size_t)CE * CD * CC * 2);
  unsigned short* ex2_wt = (unsigned short*)take((size_t)CE * CC * CD * 2);
  unsigned short* head_wt= (unsigned short*)take((size_t)CV * CC * 2);
  float*          xbuf   = (float*)take((size_t)CN * CC * 4);
  unsigned short* lnb    = (unsigned short*)take((size_t)CN * CC * 2);
  float*          qkvb   = (float*)take((size_t)CN * C3 * 4);
  float*          Qb     = (float*)take((size_t)CN * CC * 4);
  float*          Kb     = (float*)take((size_t)CN * CC * 4);
  float*          Vb     = (float*)take((size_t)CN * CC * 4);
  unsigned short* yb     = (unsigned short*)take((size_t)CN * CC * 2);
  unsigned short* hb     = (unsigned short*)take((size_t)CN * CFF * 2);
  float*          moeb   = (float*)take((size_t)CN * CC * 4);
  float*          rsc    = (float*)take((size_t)CE * CN * 4);

  auto transp = [&](const float* src, unsigned short* dst, int K_, int N_) {
    dim3 g(N_ / 32, K_ / 32);
    k_transp_bf16<<<g, 256, 0, stream>>>(src, dst, K_, N_);
  };
  auto gemm = [&](const unsigned short* A, const unsigned short* Bt, const float* bias,
                  const float* res, const float* rs, float* Cf, unsigned short* Cbo,
                  int M, int N, int K, int flags) {
    dim3 g(N / 128, M / 128);
    k_gemm_bf16_wmma<<<g, 256, 0, stream>>>(A, Bt, bias, res, rs, Cf, Cbo, M, N, K, flags);
  };

  // weight cast+transpose (recomputed every call: deterministic)
  for (int l = 0; l < CL; l++) {
    transp(qkv_w  + (size_t)l * CC * C3,  qkv_wt + (size_t)l * C3 * CC,  CC, C3);
    transp(proj_w + (size_t)l * CC * CC,  proj_wt+ (size_t)l * CC * CC,  CC, CC);
    transp(ff_w1  + (size_t)l * CC * CFF, ff1_wt + (size_t)l * CFF * CC, CC, CFF);
    transp(ff_w2  + (size_t)l * CFF * CC, ff2_wt + (size_t)l * CC * CFF, CFF, CC);
  }
  for (int e = 0; e < CE; e++) {
    transp(exp_w1 + (size_t)e * CC * CD, ex1_wt + (size_t)e * CD * CC, CC, CD);
    transp(exp_w2 + (size_t)e * CD * CC, ex2_wt + (size_t)e * CC * CD, CD, CC);
  }
  transp(head_w, head_wt, CC, CV);

  // embedding
  k_embed<<<CN, 256, 0, stream>>>(idx, tok_emb, xbuf);

  // transformer layers
  for (int l = 0; l < CL; l++) {
    k_layernorm<<<CN, 256, 0, stream>>>(xbuf, ln1_g + l * CC, ln1_b + l * CC, lnb);
    gemm(lnb, qkv_wt + (size_t)l * C3 * CC, nullptr, nullptr, nullptr,
         qkvb, nullptr, CN, C3, CC, GF_OUTF);
    k_rope_split<<<CN, 256, 0, stream>>>(qkvb, Qb, Kb, Vb);
    k_attn<<<CB * CH * (CT / 8), 256, 0, stream>>>(Qb, Kb, Vb, yb);
    gemm(yb, proj_wt + (size_t)l * CC * CC, nullptr, xbuf, nullptr,
         xbuf, nullptr, CN, CC, CC, GF_RES | GF_OUTF);                 // x += y@Wp
    k_layernorm<<<CN, 256, 0, stream>>>(xbuf, ln2_g + l * CC, ln2_b + l * CC, lnb);
    gemm(lnb, ff1_wt + (size_t)l * CFF * CC, ff_b1 + (size_t)l * CFF, nullptr, nullptr,
         nullptr, hb, CN, CFF, CC, GF_BIAS | GF_GELU | GF_OUTB);       // h = gelu(..)
    gemm(hb, ff2_wt + (size_t)l * CC * CFF, ff_b2 + (size_t)l * CC, xbuf, nullptr,
         xbuf, nullptr, CN, CC, CFF, GF_BIAS | GF_RES | GF_OUTF);      // x += h@W2+b2
  }

  // MoE: router weights, then dense scaled-accumulate experts
  k_router<<<CN / 8, 256, 0, stream>>>(xbuf, router_w, rsc);
  {
    int n = CN * CC;
    k_cvt_bf16<<<(n + 255) / 256, 256, 0, stream>>>(xbuf, lnb, n);
  }
  for (int e = 0; e < CE; e++) {
    gemm(lnb, ex1_wt + (size_t)e * CD * CC, exp_b1 + (size_t)e * CD, nullptr, nullptr,
         nullptr, hb, CN, CD, CC, GF_BIAS | GF_GELU | GF_OUTB);
    int fl = GF_BIAS | GF_SCALE | GF_OUTF | (e ? GF_ACC : 0);
    gemm(hb, ex2_wt + (size_t)e * CC * CD, exp_b2 + (size_t)e * CC, nullptr,
         rsc + (size_t)e * CN, moeb, nullptr, CN, CC, CD, fl);
  }

  // final LN + LM head
  k_layernorm<<<CN, 256, 0, stream>>>(moeb, lnf_g, lnf_b, lnb);
  gemm(lnb, head_wt, nullptr, nullptr, nullptr, logits, nullptr, CN, CV, CC, GF_OUTF);
}
